// SpeMamba_68917045232208
// MI455X (gfx1250) — matrix operand order repaired and checked by reading, not compiled, not run on hardware
//
#include <hip/hip_runtime.h>

// ---------------------------------------------------------------------------
// SpeMamba fused block for gfx1250 (MI455X).
// Kernel 1: GroupNorm statistics.  Kernel 2: fully fused block with
// v_wmma_f32_16x16x32_f16 for in-proj (4 tiles), x-proj (2 tiles), out-proj (1).
// WMMA operands come from aligned vector LDS loads of pre-transposed,
// zero-padded f16 staging buffers; SiLU uses v_rcp_f32 (no IEEE div chains).
// ---------------------------------------------------------------------------

constexpr int kB = 16, kC = 128, kH = 64, kW = 64;
constexpr int kHW = kH * kW;          // 4096
constexpr float kEps = 1e-5f;

typedef __attribute__((ext_vector_type(16))) _Float16 v16h;
typedef __attribute__((ext_vector_type(8)))  _Float16 v8h;
typedef __attribute__((ext_vector_type(8)))  float    v8f;

__device__ __forceinline__ float siluf(float v) {
  // v * sigmoid(v) with hardware rcp: v_exp + v_add + v_rcp + v_mul
  return v * __builtin_amdgcn_rcpf(1.f + __expf(-v));
}
__device__ __forceinline__ float softplusf(float v) {
  return (v > 20.f) ? v : __logf(1.f + __expf(v));
}

// ---------------------------------------------------------------------------
// Kernel 1: per-(batch,group) mean / rstd. Group (b,g) = contiguous 16384
// floats at offset (b*32+g)*16384.  stats[0..511]=mean, stats[512..1023]=rstd.
// ---------------------------------------------------------------------------
__global__ __launch_bounds__(256)
void gn_stats_kernel(const float* __restrict__ x, float* __restrict__ stats) {
  __shared__ float red[512];
  const int grp = blockIdx.x;
  const float* p = x + (size_t)grp * 16384;
  float s = 0.f, s2 = 0.f;
  for (int i = threadIdx.x; i < 16384; i += 256) {
    float v = p[i];
    s += v; s2 += v * v;
  }
  red[threadIdx.x] = s;
  red[256 + threadIdx.x] = s2;
  __syncthreads();
  for (int off = 128; off > 0; off >>= 1) {
    if (threadIdx.x < off) {
      red[threadIdx.x]       += red[threadIdx.x + off];
      red[256 + threadIdx.x] += red[256 + threadIdx.x + off];
    }
    __syncthreads();
  }
  if (threadIdx.x == 0) {
    float mean = red[0] * (1.f / 16384.f);
    float var  = red[256] * (1.f / 16384.f) - mean * mean;
    stats[grp]       = mean;
    stats[512 + grp] = rsqrtf(var + kEps);
  }
}

// ---------------------------------------------------------------------------
// Kernel 2: fused block.  256 threads (8 waves); each wave owns 2 pixels
// (16 token rows = one WMMA M tile).  Block = 16 consecutive-w pixels.
// ---------------------------------------------------------------------------
__global__ __launch_bounds__(256)
void mamba_fused_kernel(const float* __restrict__ x,
                        const float* __restrict__ gn_w, const float* __restrict__ gn_b,
                        const float* __restrict__ in_w, const float* __restrict__ conv_w,
                        const float* __restrict__ conv_b, const float* __restrict__ xproj_w,
                        const float* __restrict__ dt_w, const float* __restrict__ dt_b,
                        const float* __restrict__ A_log, const float* __restrict__ Dp,
                        const float* __restrict__ out_w, const float* __restrict__ stats,
                        float* __restrict__ out)
{
  // ---- weights staged in B-fragment layout: [n][k], f16, zero-padded K ----
  __shared__ alignas(32) _Float16 s_inwt[32 * 64];   // in_w^T   (k<16 real, k>=16 zero)
  __shared__ alignas(32) _Float16 s_bcw[32 * 32];    // xproj rows 1..32 transposed
  __shared__ alignas(32) _Float16 s_outwt[32 * 16];  // out_w^T
  __shared__ float s_dtr[32];                        // xproj row 0 (dt row), f32
  __shared__ alignas(32) float s_A[32 * 16];         // A = -exp(A_log)
  __shared__ float s_convw[32 * 4];
  __shared__ float s_convb[32], s_dtw[32], s_dtb[32], s_Dp[32];
  __shared__ float s_gnw[128], s_gnb[128];
  __shared__ float s_mu[32], s_rs[32];
  // ---- activations ----
  __shared__ float s_x[16][128];                     // raw x (residual accumulator)
  __shared__ alignas(32) _Float16 s_xn[16][128];     // groupnorm'ed x, f16
  __shared__ alignas(32) _Float16 s_xc[8][16][32];   // per wave: xc, then y in place
  __shared__ alignas(32) _Float16 s_zs[8][16][32];   // per wave: silu(z)
  __shared__ alignas(32) _Float16 s_bc[8][2][8][32]; // per wave: [pix][tok][B(16)|C(16)]
  __shared__ float s_dt[8][16];                      // per wave: raw dt per row

  const int tid = threadIdx.x;
  const int pix_base = blockIdx.x * 16;
  const int b  = pix_base / kHW;
  const int hw = pix_base % kHW;
  const size_t xbase = (size_t)b * kC * kHW + hw;    // + c*kHW + p

  // ---------------- phase A: stage everything from global ----------------
  for (int i = tid; i < 32 * 64; i += 256) {         // in_w^T, K-padded
    int e = i >> 5, k = i & 31;
    s_inwt[i] = (k < 16) ? (_Float16)in_w[e * 16 + k] : (_Float16)0.f;
  }
  for (int i = tid; i < 32 * 32; i += 256) {         // xproj rows 1..32, transposed
    int n = i >> 5, k = i & 31;                      // n<16 -> B col n, n>=16 -> C col
    s_bcw[i] = (_Float16)xproj_w[(1 + n) * 32 + k];
  }
  for (int i = tid; i < 32 * 16; i += 256) {         // out_w^T
    int n = i >> 5, k = i & 31;
    s_outwt[i] = (_Float16)out_w[n * 32 + k];
  }
  for (int i = tid; i < 32 * 16; i += 256) s_A[i] = -__expf(A_log[i]);
  if (tid < 128) {
    s_convw[tid] = conv_w[tid];
    s_gnw[tid]   = gn_w[tid];
    s_gnb[tid]   = gn_b[tid];
  }
  if (tid < 32) {
    s_dtr[tid]   = xproj_w[tid];
    s_convb[tid] = conv_b[tid];
    s_dtw[tid]   = dt_w[tid];
    s_dtb[tid]   = dt_b[tid];
    s_Dp[tid]    = Dp[tid];
    s_mu[tid]    = stats[b * 32 + tid];
    s_rs[tid]    = stats[512 + b * 32 + tid];
  }
  for (int i = tid; i < 2048; i += 256) {            // raw x tile, coalesced
    int c = i >> 4, p = i & 15;
    s_x[p][c] = x[xbase + (size_t)c * kHW + p];
  }
  __syncthreads();

  // ---------------- phase B: groupnorm -> f16 staged copy -----------------
  for (int i = tid; i < 2048; i += 256) {
    int c = i >> 4, p = i & 15;
    int g = c >> 2;
    float v = (s_x[p][c] - s_mu[g]) * s_rs[g] * s_gnw[c] + s_gnb[c];
    s_xn[p][c] = (_Float16)v;
  }
  __syncthreads();

  const int wv   = tid >> 5;
  const int lane = tid & 31;
  const int hi   = lane >> 4;
  const int ln   = lane & 15;
  const int p0   = wv * 2;
  const int kb   = ln + 16 * hi;
  const v8f zero8 = {};
  const v8h zeroh = {};

  // ---- matmul 1: xz = seq(16x16) @ in_w^T (K padded to 32 via zeroed B) ----
  v16h a1;
  {
    const int m = ln, pm = p0 + (m >> 3), tok = m & 7;
    const v16h z16 = {};
    a1 = hi ? z16 : *(const v16h*)&s_xn[pm][tok * 16];
  }
  v8f acc[4];
#pragma unroll
  for (int j = 0; j < 4; ++j) {
    v16h bf = *(const v16h*)&s_inwt[(16 * j + ln) * 32 + 16 * hi];
    acc[j] = __builtin_amdgcn_wmma_f32_16x16x32_f16(false, a1, false, bf,
                                                    (short)0, zero8, false, false);
  }

  // ---- causal depthwise conv + SiLU, in-lane over the v8f token axis ----
#pragma unroll
  for (int t2 = 0; t2 < 2; ++t2) {
    const int d = t2 * 16 + ln;
    const float w0c = s_convw[d * 4 + 0], w1c = s_convw[d * 4 + 1];
    const float w2c = s_convw[d * 4 + 2], w3c = s_convw[d * 4 + 3];
    const float cb  = s_convb[d];
    float xi[8];
#pragma unroll
    for (int r = 0; r < 8; ++r) xi[r] = acc[t2][r];
#pragma unroll
    for (int r = 0; r < 8; ++r) {
      float a = cb + xi[r] * w3c;
      if (r >= 1) a += xi[r - 1] * w2c;
      if (r >= 2) a += xi[r - 2] * w1c;
      if (r >= 3) a += xi[r - 3] * w0c;
      const int m = hi * 8 + r;
      s_xc[wv][m][d] = (_Float16)siluf(a);
      s_zs[wv][m][d] = (_Float16)siluf(acc[2 + t2][r]);
    }
  }
  __syncthreads();

  // ---- dt scalar per row: branch-free half-dot + xor-shuffle combine ----
  {
    v8h c0 = *(const v8h*)&s_xc[wv][ln][16 * hi];
    v8h c1 = *(const v8h*)&s_xc[wv][ln][16 * hi + 8];
    float part = 0.f;
#pragma unroll
    for (int q = 0; q < 8; ++q)
      part += (float)c0[q] * s_dtr[16 * hi + q] +
              (float)c1[q] * s_dtr[16 * hi + 8 + q];
    float tot = part + __shfl_xor(part, 16);
    s_dt[wv][ln] = tot;                 // lanes L and L^16 store identical value
  }

  // ---- matmul 2: [B|C] = xc(16x32) @ xproj_w[1:33]^T ----
  v16h a2;
  {
    a2 = *(const v16h*)&s_xc[wv][ln][16 * hi];
  }
  v16h bB = *(const v16h*)&s_bcw[ln * 32 + 16 * hi];
  v16h bC = *(const v16h*)&s_bcw[(16 + ln) * 32 + 16 * hi];
  v8f accB = __builtin_amdgcn_wmma_f32_16x16x32_f16(false, a2, false, bB,
                                                    (short)0, zero8, false, false);
  v8f accC = __builtin_amdgcn_wmma_f32_16x16x32_f16(false, a2, false, bC,
                                                    (short)0, zero8, false, false);
#pragma unroll
  for (int r = 0; r < 8; ++r) {
    s_bc[wv][hi][r][ln]      = (_Float16)accB[r];
    s_bc[wv][hi][r][16 + ln] = (_Float16)accC[r];
  }
  __syncthreads();

  // ---- selective scan: lane = d (0..31), both pixels, state in registers ----
  {
    const int d = lane;
    v8f A0 = *(const v8f*)&s_A[d * 16];
    v8f A1 = *(const v8f*)&s_A[d * 16 + 8];
    const float dtw = s_dtw[d], dtb = s_dtb[d], dpv = s_Dp[d];
    float hst[2][16];
#pragma unroll
    for (int p = 0; p < 2; ++p)
#pragma unroll
      for (int s = 0; s < 16; ++s) hst[p][s] = 0.f;

    for (int t = 0; t < 8; ++t) {
#pragma unroll
      for (int p = 0; p < 2; ++p) {
        const int m = p * 8 + t;
        const float dtf = softplusf(s_dt[wv][m] * dtw + dtb);
        const float xt  = (float)s_xc[wv][m][d];
        const float dx  = dtf * xt;
        v16h Brow = *(const v16h*)&s_bc[wv][p][t][0];   // broadcast loads
        v16h Crow = *(const v16h*)&s_bc[wv][p][t][16];
        float ysum = 0.f;
#pragma unroll
        for (int s = 0; s < 16; ++s) {
          const float Av = (s < 8) ? A0[s & 7] : A1[s & 7];
          const float dA = __expf(dtf * Av);
          hst[p][s] = dA * hst[p][s] + dx * (float)Brow[s];
          ysum += hst[p][s] * (float)Crow[s];
        }
        const float y = (ysum + xt * dpv) * (float)s_zs[wv][m][d];
        s_xc[wv][m][d] = (_Float16)y;   // y replaces xc in place
      }
    }
  }
  __syncthreads();

  // ---- matmul 3: out = y(16x32) @ out_w^T ----
  v16h a3;
  {
    a3 = *(const v16h*)&s_xc[wv][ln][16 * hi];
  }
  v16h bO = *(const v16h*)&s_outwt[ln * 32 + 16 * hi];
  v8f accO = __builtin_amdgcn_wmma_f32_16x16x32_f16(false, a3, false, bO,
                                                    (short)0, zero8, false, false);

  // residual add in place (each (pixel,c) touched by exactly one lane)
#pragma unroll
  for (int r = 0; r < 8; ++r) {
    const int c = r * 16 + ln;
    s_x[p0 + hi][c] += accO[r];
  }
  __syncthreads();

  // coalesced store
  for (int i = tid; i < 2048; i += 256) {
    int c = i >> 4, p = i & 15;
    out[xbase + (size_t)c * kHW + p] = s_x[p][c];
  }
}

extern "C" void kernel_launch(void* const* d_in, const int* in_sizes, int n_in,
                              void* d_out, int out_size, void* d_ws, size_t ws_size,
                              hipStream_t stream) {
  (void)in_sizes; (void)n_in; (void)out_size; (void)ws_size;
  const float* x       = (const float*)d_in[0];
  const float* gn_w    = (const float*)d_in[1];
  const float* gn_b    = (const float*)d_in[2];
  const float* in_w    = (const float*)d_in[3];
  const float* conv_w  = (const float*)d_in[4];
  const float* conv_b  = (const float*)d_in[5];
  const float* xproj_w = (const float*)d_in[6];
  const float* dt_w    = (const float*)d_in[7];
  const float* dt_b    = (const float*)d_in[8];
  const float* A_log   = (const float*)d_in[9];
  const float* Dp      = (const float*)d_in[10];
  const float* out_w   = (const float*)d_in[11];
  float* stats = (float*)d_ws;                       // 1024 floats
  float* out   = (float*)d_out;

  gn_stats_kernel<<<kB * 32, 256, 0, stream>>>(x, stats);
  mamba_fused_kernel<<<(kB * kHW) / 16, 256, 0, stream>>>(
      x, gn_w, gn_b, in_w, conv_w, conv_b, xproj_w, dt_w, dt_b, A_log, Dp,
      out_w, stats, out);
}
